// MobileCrossViTBlockv2_81192061763898
// MI455X (gfx1250) — matrix operand order repaired and checked by reading, not compiled, not run on hardware
//
#include <hip/hip_runtime.h>
#include <hip/hip_bf16.h>
#include <math.h>

typedef __bf16 bf16_t;
typedef __bf16 v16bf __attribute__((ext_vector_type(16)));
typedef __bf16 v8bf  __attribute__((ext_vector_type(8)));
typedef __bf16 v2bf  __attribute__((ext_vector_type(2)));
typedef float  v8f   __attribute__((ext_vector_type(8)));
typedef float  v4f   __attribute__((ext_vector_type(4)));

#define NB    16     // batch
#define AC    256    // attention channels
#define NPOS  4096   // 4 patch-positions * 1024 patches
#define HW    64

__device__ __forceinline__ float gelu_f(float x){
    return 0.5f * x * (1.0f + erff(x * 0.7071067811865476f));
}

enum { EPI_BIAS_F32 = 0, EPI_RELU_CTX = 1, EPI_BIAS_RES = 2, EPI_GELU_BF16 = 3, EPI_BN_GELU_FOLD = 4 };
enum { B_PLAIN = 0, B_DIFF = 1 };

// ---------------------------------------------------------------------------
// Tiled bf16 WMMA GEMM:  D[b] = W (Cout x Ktot, fp32) * Act[b] (Ktot x 4096, bf16)
// Output tile 128x128 per 256-thread block; 8 waves, each 32(M) x 64(N);
// per-wave 2x4 = 8 WMMA accumulators of 16x16.
// ---------------------------------------------------------------------------
template<int EPI, int BLD>
__global__ __launch_bounds__(256)
void gemm_kernel(const float* __restrict__ W,
                 const float* __restrict__ bias,    // bias (or BN gamma for FOLD)
                 const float* __restrict__ bias2,   // BN beta for FOLD
                 const bf16_t* __restrict__ act0,
                 const bf16_t* __restrict__ act1,   // second source for B_DIFF
                 const float* __restrict__ ctxp,    // [NB][Cout][4] for RELU_CTX
                 const float* __restrict__ res,     // residual fp32
                 float* __restrict__ outf,
                 bf16_t* __restrict__ outb,
                 int Cout, int Ktot, int actChan)
{
    constexpr int BM = 128, BN = 128, BK = 32, LDST = 40;  // 40*2B row stride: 16B aligned, conflict-free frags
    __shared__ bf16_t ldsA[BM * LDST];
    __shared__ bf16_t ldsB[BN * LDST];

    const int b       = blockIdx.z;
    const int posBase = blockIdx.x * BN;
    const int coBase  = blockIdx.y * BM;
    const int tid  = threadIdx.x;
    const int lane = tid & 31;
    const int w    = tid >> 5;
    const int wm   = w >> 1;          // 0..3 -> M offset wm*32
    const int wn   = w & 1;           // 0..1 -> N offset wn*64
    const int lr   = lane & 15;
    const int lh   = lane >> 4;

    v8f acc[2][4];
    #pragma unroll
    for (int i = 0; i < 2; i++)
        #pragma unroll
        for (int j = 0; j < 4; j++)
            #pragma unroll
            for (int e = 0; e < 8; e++) acc[i][j][e] = 0.0f;

    // A loader: thread -> (row, 16-float chunk)
    const int arow = tid >> 1;
    const int acol = (tid & 1) * 16;
    // B loader: thread -> 2 ci rows x 8 positions
    const int bci  = (tid >> 4) * 2;      // 0..30
    const int bpos = (tid & 15) * 8;      // 0..120

    const long actBase = (long)b * actChan * NPOS + posBase + bpos;

    for (int k0 = 0; k0 < Ktot; k0 += BK) {
        // ---- A tile: fp32 weights -> bf16 LDS [co][k] ----
        {
            const float* src = W + (long)(coBase + arow) * Ktot + k0 + acol;
            v4f f0 = *(const v4f*)(src);
            v4f f1 = *(const v4f*)(src + 4);
            v4f f2 = *(const v4f*)(src + 8);
            v4f f3 = *(const v4f*)(src + 12);
            bf16_t t16[16];
            #pragma unroll
            for (int j = 0; j < 4; j++) {
                t16[j]      = (bf16_t)f0[j];
                t16[4 + j]  = (bf16_t)f1[j];
                t16[8 + j]  = (bf16_t)f2[j];
                t16[12 + j] = (bf16_t)f3[j];
            }
            const int basee = arow * LDST + acol;
            #pragma unroll
            for (int j = 0; j < 8; j++) {
                v2bf pr; pr[0] = t16[2 * j]; pr[1] = t16[2 * j + 1];
                *(v2bf*)&ldsA[basee + 2 * j] = pr;
            }
        }
        // ---- B tile: bf16 activations [k][pos] -> transposed LDS [pos][k] ----
        {
            v8bf r0, r1;
            bool plainPart = (BLD == B_PLAIN) || (k0 < 256);
            if (plainPart) {
                const bf16_t* s0 = act0 + actBase + (long)(k0 + bci) * NPOS;
                r0 = *(const v8bf*)(s0);
                r1 = *(const v8bf*)(s0 + NPOS);
            } else {
                const bf16_t* s0 = act0 + actBase + (long)(k0 - 256 + bci) * NPOS;
                const bf16_t* s1 = act1 + actBase + (long)(k0 - 256 + bci) * NPOS;
                v8bf a0 = *(const v8bf*)(s0), a1 = *(const v8bf*)(s0 + NPOS);
                v8bf c0 = *(const v8bf*)(s1), c1 = *(const v8bf*)(s1 + NPOS);
                #pragma unroll
                for (int j = 0; j < 8; j++) {
                    r0[j] = (bf16_t)((float)a0[j] - (float)c0[j]);
                    r1[j] = (bf16_t)((float)a1[j] - (float)c1[j]);
                }
            }
            #pragma unroll
            for (int j = 0; j < 8; j++) {
                v2bf pr; pr[0] = r0[j]; pr[1] = r1[j];
                *(v2bf*)&ldsB[(bpos + j) * LDST + bci] = pr;
            }
        }
        __syncthreads();

        // ---- fragments + WMMA ----
        v16bf af[2], bfr[4];
        #pragma unroll
        for (int mi = 0; mi < 2; mi++) {
            const int row = wm * 32 + mi * 16 + lr;
            const int kb  = lh * 8;                      // A: K = {kb..kb+7, kb+16..kb+23}
            v8bf lo = *(const v8bf*)&ldsA[row * LDST + kb];
            v8bf hi = *(const v8bf*)&ldsA[row * LDST + kb + 16];
            af[mi] = __builtin_shufflevector(lo, hi, 0,1,2,3,4,5,6,7,8,9,10,11,12,13,14,15);
        }
        #pragma unroll
        for (int ni = 0; ni < 4; ni++) {
            const int n  = wn * 64 + ni * 16 + lr;
            const int kb = lh * 8;                       // B: K = {kb..kb+7, kb+16..kb+23}
            v8bf lo = *(const v8bf*)&ldsB[n * LDST + kb];
            v8bf hi = *(const v8bf*)&ldsB[n * LDST + kb + 16];
            bfr[ni] = __builtin_shufflevector(lo, hi, 0,1,2,3,4,5,6,7,8,9,10,11,12,13,14,15);
        }
        #pragma unroll
        for (int mi = 0; mi < 2; mi++)
            #pragma unroll
            for (int ni = 0; ni < 4; ni++)
                acc[mi][ni] = __builtin_amdgcn_wmma_f32_16x16x32_bf16(
                    false, af[mi], false, bfr[ni], (short)0, acc[mi][ni], false, false);
        __syncthreads();
    }

    // ---- epilogue ----
    #pragma unroll
    for (int mi = 0; mi < 2; mi++) {
        #pragma unroll
        for (int ni = 0; ni < 4; ni++) {
            #pragma unroll
            for (int i = 0; i < 8; i++) {
                const int co  = coBase + wm * 32 + mi * 16 + lh * 8 + i;
                const int pos = posBase + wn * 64 + ni * 16 + lr;
                float v = acc[mi][ni][i];
                const long oidx = ((long)b * Cout + co) * NPOS + pos;
                if (EPI == EPI_BIAS_F32) {
                    outf[oidx] = v + bias[co];
                } else if (EPI == EPI_RELU_CTX) {
                    const float c = ctxp[((long)b * Cout + co) * 4 + (pos >> 10)];
                    const float r = fmaxf(v + bias[co], 0.0f);
                    outb[oidx] = (bf16_t)(r * c);
                } else if (EPI == EPI_BIAS_RES) {
                    outf[oidx] = v + bias[co] + res[oidx];
                } else if (EPI == EPI_GELU_BF16) {
                    outb[oidx] = (bf16_t)gelu_f(v + bias[co]);
                } else if (EPI == EPI_BN_GELU_FOLD) {
                    const float sc = bias[co] * rsqrtf(1.0f + 1e-5f);
                    const float y  = gelu_f(v * sc + bias2[co]);
                    const int p = pos >> 10, n = pos & 1023;
                    const int hh = (n >> 5) * 2 + (p >> 1);
                    const int ww = (n & 31) * 2 + (p & 1);
                    outf[(((long)b * Cout + co) * HW + hh) * HW + ww] = y;
                }
            }
        }
    }
}

// ---------------------------------------------------------------------------
// Small helper kernels
// ---------------------------------------------------------------------------
__global__ void zero_kernel(float* p, int n){
    int i = blockIdx.x * blockDim.x + threadIdx.x;
    if (i < n) p[i] = 0.0f;
}

// per-sample sum / sumsq over 256*4096 elements; grid (64, NB)
__global__ __launch_bounds__(256) void stats_kernel(const float* __restrict__ x, float* __restrict__ st){
    const int b = blockIdx.y;
    const long base = (long)b * (AC * NPOS) + (long)blockIdx.x * 16384 + threadIdx.x;
    float s = 0.0f, q = 0.0f;
    #pragma unroll 4
    for (int i = 0; i < 64; i++) { float v = x[base + (long)i * 256]; s += v; q += v * v; }
    __shared__ float ls[256], lq[256];
    ls[threadIdx.x] = s; lq[threadIdx.x] = q; __syncthreads();
    for (int o = 128; o > 0; o >>= 1) {
        if (threadIdx.x < (unsigned)o) { ls[threadIdx.x] += ls[threadIdx.x + o]; lq[threadIdx.x] += lq[threadIdx.x + o]; }
        __syncthreads();
    }
    if (threadIdx.x == 0) { atomicAdd(&st[b * 2], ls[0]); atomicAdd(&st[b * 2 + 1], lq[0]); }
}

// GroupNorm(num_groups=1) + fp32->bf16; grid NB*AC blocks of one (b,c) row
__global__ __launch_bounds__(256) void gn_prep_kernel(const float* __restrict__ x, const float* __restrict__ st,
        const float* __restrict__ g, const float* __restrict__ bt, bf16_t* __restrict__ out){
    const int bc = blockIdx.x;
    const int b = bc >> 8, c = bc & 255;
    const float inv = 1.0f / (256.0f * 4096.0f);
    const float m  = st[b * 2] * inv;
    const float vr = st[b * 2 + 1] * inv - m * m;
    const float rs = rsqrtf(vr + 1e-5f);
    const float sc = g[c] * rs;
    const float sh = bt[c] - m * sc;
    const long base = (long)bc * NPOS;
    for (int i = threadIdx.x; i < NPOS; i += 256)
        out[base + i] = (bf16_t)(x[base + i] * sc + sh);
}

// fp32 [b][128][64][64] -> bf16 unfold layout [b][128][4][1024]; grid 32768
__global__ __launch_bounds__(256) void unfold_prep_kernel(const float* __restrict__ x, bf16_t* __restrict__ out){
    const long i = (long)blockIdx.x * 256 + threadIdx.x;
    const int n = (int)(i & 1023);
    long t = i >> 10;
    const int p = (int)(t & 3); t >>= 2;                 // t = b*128 + c
    const int hh = (n >> 5) * 2 + (p >> 1);
    const int ww = (n & 31) * 2 + (p & 1);
    out[i] = (bf16_t)x[(t * HW + hh) * HW + ww];
}

// q = q_w . xO + q_b, softmax over n; one block per (b,p) -> score[b][p][1024]
__global__ __launch_bounds__(256) void qsoftmax_kernel(const float* __restrict__ xo, const float* __restrict__ qw,
        const float* __restrict__ qb, float* __restrict__ score){
    const int bp = blockIdx.x, b = bp >> 2, p = bp & 3;
    __shared__ float red[256];
    const long base = (long)b * AC * NPOS + (long)p * 1024;
    const int t = threadIdx.x;
    float q0 = qb[0], q1 = q0, q2 = q0, q3 = q0;
    for (int ci = 0; ci < AC; ci++) {
        const float wv = qw[ci];
        const float* row = xo + base + (long)ci * NPOS;
        q0 += wv * row[t]; q1 += wv * row[t + 256]; q2 += wv * row[t + 512]; q3 += wv * row[t + 768];
    }
    float mx = fmaxf(fmaxf(q0, q1), fmaxf(q2, q3));
    red[t] = mx; __syncthreads();
    for (int o = 128; o > 0; o >>= 1) { if (t < o) red[t] = fmaxf(red[t], red[t + o]); __syncthreads(); }
    mx = red[0]; __syncthreads();
    const float e0 = expf(q0 - mx), e1 = expf(q1 - mx), e2 = expf(q2 - mx), e3 = expf(q3 - mx);
    red[t] = e0 + e1 + e2 + e3; __syncthreads();
    for (int o = 128; o > 0; o >>= 1) { if (t < o) red[t] += red[t + o]; __syncthreads(); }
    const float ri = 1.0f / red[0];
    float* so = score + (long)bp * 1024;
    so[t] = e0 * ri; so[t + 256] = e1 * ri; so[t + 512] = e2 * ri; so[t + 768] = e3 * ri;
}

// hs[b][ci][p] = sum_n h[b][ci][p][n] * score[b][p][n]; grid NB*AC*4, block 128
__global__ __launch_bounds__(128) void hs_kernel(const bf16_t* __restrict__ h, const float* __restrict__ score,
        float* __restrict__ hs){
    const int g = blockIdx.x;
    const int p = g & 3, ci = (g >> 2) & 255, b = g >> 10;
    const bf16_t* row = h + ((long)(b * AC + ci)) * NPOS + (long)p * 1024;
    const float* sr = score + ((long)(b * 4 + p)) * 1024;
    const int t = threadIdx.x;
    float s = 0.0f;
    #pragma unroll
    for (int j = 0; j < 8; j++) { const int n = t + j * 128; s += (float)row[n] * sr[n]; }
    __shared__ float red[128];
    red[t] = s; __syncthreads();
    for (int o = 64; o > 0; o >>= 1) { if (t < o) red[t] += red[t + o]; __syncthreads(); }
    if (t == 0) hs[(long)(b * AC + ci) * 4 + p] = red[0];
}

// ctx[b][co][p] = k_w . hs + k_b  (softmax weights sum to 1, so bias is exact)
__global__ __launch_bounds__(256) void ctx_kernel(const float* __restrict__ kw, const float* __restrict__ kb,
        const float* __restrict__ hs, float* __restrict__ ctx){
    const int b = blockIdx.x, co = threadIdx.x;
    float a0 = kb[co], a1 = a0, a2 = a0, a3 = a0;
    const float* hb = hs + (long)b * AC * 4;
    const float* wr = kw + (long)co * AC;
    for (int ci = 0; ci < AC; ci++) {
        const float wv = wr[ci];
        const float* hp = hb + ci * 4;
        a0 += wv * hp[0]; a1 += wv * hp[1]; a2 += wv * hp[2]; a3 += wv * hp[3];
    }
    float* cp = ctx + ((long)b * AC + co) * 4;
    cp[0] = a0; cp[1] = a1; cp[2] = a2; cp[3] = a3;
}

// ---------------------------------------------------------------------------
extern "C" void kernel_launch(void* const* d_in, const int* in_sizes, int n_in,
                              void* d_out, int out_size, void* d_ws, size_t ws_size,
                              hipStream_t stream) {
    (void)in_sizes; (void)n_in; (void)out_size; (void)ws_size;
    const float* x1_in    = (const float*)d_in[0];
    const float* x2_in    = (const float*)d_in[1];
    const float* local1_w = (const float*)d_in[2];
    const float* local1_b = (const float*)d_in[3];
    const float* local2_w = (const float*)d_in[4];
    const float* local2_b = (const float*)d_in[5];
    const float* gn1_g = (const float*)d_in[70];
    const float* gn1_b = (const float*)d_in[71];
    const float* gn2_g = (const float*)d_in[72];
    const float* gn2_b = (const float*)d_in[73];
    const float* proj1_w    = (const float*)d_in[74];
    const float* proj1_bn_g = (const float*)d_in[75];
    const float* proj1_bn_b = (const float*)d_in[76];
    const float* proj2_w    = (const float*)d_in[77];
    const float* proj2_bn_g = (const float*)d_in[78];
    const float* proj2_bn_b = (const float*)d_in[79];

    // workspace layout (~269 MB)
    float*  X1   = (float*)d_ws;                                   // NB*256*4096 fp32
    float*  X2   = X1 + (size_t)NB * AC * NPOS;
    bf16_t* hbuf = (bf16_t*)(X2 + (size_t)NB * AC * NPOS);         // NB*512*4096 bf16 capacity
    bf16_t* ubuf = hbuf + (size_t)NB * 512 * NPOS;
    float*  score = (float*)(ubuf + (size_t)NB * 512 * NPOS);      // NB*4*1024
    float*  st   = score + NB * 4 * 1024;                          // 32 (+pad)
    float*  ctx  = st + 64;                                        // NB*256*4
    float*  hs   = ctx + NB * AC * 4;                              // NB*256*4

    float* y1 = (float*)d_out;
    float* y2 = y1 + (size_t)NB * 128 * HW * HW;

    const dim3 blk(256);

    // local 1x1 convs (128 -> 256) in unfold layout
    unfold_prep_kernel<<<32768, blk, 0, stream>>>(x1_in, hbuf);
    gemm_kernel<EPI_BIAS_F32, B_PLAIN><<<dim3(32, 2, NB), blk, 0, stream>>>(
        local1_w, local1_b, nullptr, hbuf, nullptr, nullptr, nullptr, X1, nullptr, 256, 128, 128);
    unfold_prep_kernel<<<32768, blk, 0, stream>>>(x2_in, hbuf);
    gemm_kernel<EPI_BIAS_F32, B_PLAIN><<<dim3(32, 2, NB), blk, 0, stream>>>(
        local2_w, local2_b, nullptr, hbuf, nullptr, nullptr, nullptr, X2, nullptr, 256, 128, 128);

    // attn+ffn steps: blk1[0](X1,X2), blk2[0](X2,X1), blk1[1], blk2[1]
    const int stepBase[4] = {6, 38, 22, 54};
    for (int s = 0; s < 4; s++) {
        float* T = (s & 1) ? X2 : X1;
        const float* O = (s & 1) ? X1 : X2;
        const int bs = stepBase[s];
        const float* gn_g  = (const float*)d_in[bs + 0];
        const float* gn_b  = (const float*)d_in[bs + 1];
        const float* q_w   = (const float*)d_in[bs + 2];
        const float* q_b   = (const float*)d_in[bs + 3];
        const float* k_w   = (const float*)d_in[bs + 4];
        const float* k_b   = (const float*)d_in[bs + 5];
        const float* v_w   = (const float*)d_in[bs + 6];
        const float* v_b   = (const float*)d_in[bs + 7];
        const float* o_w   = (const float*)d_in[bs + 8];
        const float* o_b   = (const float*)d_in[bs + 9];
        const float* fgn_g = (const float*)d_in[bs + 10];
        const float* fgn_b = (const float*)d_in[bs + 11];
        const float* f1_w  = (const float*)d_in[bs + 12];
        const float* f1_b  = (const float*)d_in[bs + 13];
        const float* f2_w  = (const float*)d_in[bs + 14];
        const float* f2_b  = (const float*)d_in[bs + 15];

        // h = GN(T) -> bf16
        zero_kernel<<<1, 64, 0, stream>>>(st, 64);
        stats_kernel<<<dim3(64, NB), blk, 0, stream>>>(T, st);
        gn_prep_kernel<<<NB * AC, blk, 0, stream>>>(T, st, gn_g, gn_b, hbuf);
        // score = softmax(q_w . O + q_b) over n
        qsoftmax_kernel<<<NB * 4, blk, 0, stream>>>(O, q_w, q_b, score);
        // ctx = k_w . (h . score) + k_b   (exact; softmax sums to 1)
        hs_kernel<<<NB * AC * 4, 128, 0, stream>>>(hbuf, score, hs);
        ctx_kernel<<<NB, blk, 0, stream>>>(k_w, k_b, hs, ctx);
        // u = relu(v_w.h + v_b) * ctx  -> bf16
        gemm_kernel<EPI_RELU_CTX, B_PLAIN><<<dim3(32, 2, NB), blk, 0, stream>>>(
            v_w, v_b, nullptr, hbuf, nullptr, ctx, nullptr, nullptr, ubuf, 256, 256, 256);
        // T = o_w.u + o_b + T  (in place)
        gemm_kernel<EPI_BIAS_RES, B_PLAIN><<<dim3(32, 2, NB), blk, 0, stream>>>(
            o_w, o_b, nullptr, ubuf, nullptr, nullptr, T, T, nullptr, 256, 256, 256);
        // FFN: g = gelu(f1_w.GN(T)+f1_b); T += f2_w.g + f2_b
        zero_kernel<<<1, 64, 0, stream>>>(st, 64);
        stats_kernel<<<dim3(64, NB), blk, 0, stream>>>(T, st);
        gn_prep_kernel<<<NB * AC, blk, 0, stream>>>(T, st, fgn_g, fgn_b, hbuf);
        gemm_kernel<EPI_GELU_BF16, B_PLAIN><<<dim3(32, 4, NB), blk, 0, stream>>>(
            f1_w, f1_b, nullptr, hbuf, nullptr, nullptr, nullptr, nullptr, ubuf, 512, 256, 256);
        gemm_kernel<EPI_BIAS_RES, B_PLAIN><<<dim3(32, 2, NB), blk, 0, stream>>>(
            f2_w, f2_b, nullptr, ubuf, nullptr, nullptr, T, T, nullptr, 256, 512, 512);
    }

    // final GN on both streams -> bf16 g1 (hbuf), g2 (ubuf)
    zero_kernel<<<1, 64, 0, stream>>>(st, 64);
    stats_kernel<<<dim3(64, NB), blk, 0, stream>>>(X1, st);
    gn_prep_kernel<<<NB * AC, blk, 0, stream>>>(X1, st, gn1_g, gn1_b, hbuf);
    zero_kernel<<<1, 64, 0, stream>>>(st, 64);
    stats_kernel<<<dim3(64, NB), blk, 0, stream>>>(X2, st);
    gn_prep_kernel<<<NB * AC, blk, 0, stream>>>(X2, st, gn2_g, gn2_b, ubuf);

    // y1 = gelu(BN(proj1_w . [g1; g1-g2])), folded to [b][128][64][64]
    gemm_kernel<EPI_BN_GELU_FOLD, B_DIFF><<<dim3(32, 1, NB), blk, 0, stream>>>(
        proj1_w, proj1_bn_g, proj1_bn_b, hbuf, ubuf, nullptr, nullptr, y1, nullptr, 128, 512, 256);
    // y2 = gelu(BN(proj2_w . [g2; g2-g1]))
    gemm_kernel<EPI_BN_GELU_FOLD, B_DIFF><<<dim3(32, 1, NB), blk, 0, stream>>>(
        proj2_w, proj2_bn_g, proj2_bn_b, ubuf, hbuf, nullptr, nullptr, y2, nullptr, 128, 512, 256);
}
